// Model_11012296147422
// MI455X (gfx1250) — matrix operand order repaired and checked by reading, compiled, not run on hardware
//
#include <hip/hip_runtime.h>
#include <hip/hip_bf16.h>

// ---------------- problem constants ----------------
#define Bx    128
#define Pn    168
#define Mout  8            // M1*M2*M3
#define HIDC  50
#define HIDR  50
#define HIDS  5
#define CK    6
#define SKIP  24
#define HW    24
#define PT    6            // (P-CK)//SKIP
#define Lt    163          // P-CK+1
#define Kdim  1344         // P*M
#define Ntot  50400        // P*HIDC*CK
#define KC    96           // K chunk held in LDS (3 wmma k-steps)
#define NT    32           // N tile per workgroup (2 x 16-col subtiles)

typedef __attribute__((ext_vector_type(16))) __bf16 v16bf;
typedef __attribute__((ext_vector_type(8)))  float  v8f;

// ---------------- 1) x fp32 -> bf16 (done once per call; stays L2-hot) ----
__global__ void cvt_x_kernel(const float* __restrict__ x, __bf16* __restrict__ xbf) {
    int i = blockIdx.x * blockDim.x + threadIdx.x;
    if (i < Bx * Kdim) xbf[i] = (__bf16)x[i];
}

// ---------------- 2) WMMA GEMM: c = relu(xf @ conv_w.T + conv_b) ----------
// grid: Ntot/NT = 1575 workgroups, block = 256 threads (8 waves).
// Each wave: 16 rows of M (8*16 = 128 = all batches) x 32 cols of N.
// conv_w streamed fp32->bf16 through LDS (vectorized b128 fill), read once.
__global__ void __launch_bounds__(256)
conv_gemm_kernel(const __bf16* __restrict__ xbf,
                 const float*  __restrict__ conv_w,
                 const float*  __restrict__ conv_b,
                 float* __restrict__ c) {
    __shared__ __bf16 Bs[NT][KC];   // column-contiguous K for ds_load_b128 frags

    const int n0   = blockIdx.x * NT;
    const int lane = threadIdx.x & 31;
    const int wave = threadIdx.x >> 5;
    const int col  = lane & 15;
    const int hi   = lane >> 4;                  // lane half (K interleave)
    const int row  = wave * 16 + col;            // A-fragment row for this lane

    union Frag { v16bf v; uint4 u[2]; };
    union Pk4  { __bf16 h[4]; uint2 u; };

    // fill-loop mapping is invariant across K chunks: hoist address math.
    // NT*KC/4 = 768 float4 transfers, 3 per thread.
    size_t  src_off[3];
    __bf16* dst_ptr[3];
#pragma unroll
    for (int j = 0; j < 3; ++j) {
        int i  = threadIdx.x + j * 256;
        int bc = i / (KC / 4);                   // column 0..31
        int kq = i - bc * (KC / 4);              // float4 index 0..23
        src_off[j] = (size_t)(n0 + bc) * Kdim + kq * 4;
        dst_ptr[j] = &Bs[bc][kq * 4];
    }

    v8f acc0 = {};
    v8f acc1 = {};
    const __bf16* arow = xbf + (size_t)row * Kdim;

    for (int kc0 = 0; kc0 < Kdim; kc0 += KC) {
        __syncthreads();
        // issue the 3 independent b128 conv_w loads, then convert+store bf16x4
        float4 f[3];
#pragma unroll
        for (int j = 0; j < 3; ++j)
            f[j] = *(const float4*)(conv_w + src_off[j] + kc0);
#pragma unroll
        for (int j = 0; j < 3; ++j) {
            Pk4 pk;
            pk.h[0] = (__bf16)f[j].x; pk.h[1] = (__bf16)f[j].y;
            pk.h[2] = (__bf16)f[j].z; pk.h[3] = (__bf16)f[j].w;
            *(uint2*)dst_ptr[j] = pk.u;
        }
        // prefetch next chunk: 32 rows x 3 cachelines (96 floats) each
        if (kc0 + KC < Kdim && threadIdx.x < NT * 3) {
            __builtin_prefetch(conv_w + (size_t)(n0 + threadIdx.x / 3) * Kdim
                                      + kc0 + KC + (threadIdx.x % 3) * 32, 0, 1);
        }
        __syncthreads();

        // preload all 9 fragments of this chunk, then 6 back-to-back WMMAs
        Frag a[3], b0[3], b1[3];
#pragma unroll
        for (int s = 0; s < 3; ++s) {
            const int kk = s * 32;
            // A frag: lanes 0-15 hold K {0-15}, lanes 16-31 hold {16-31}
            a[s].u[0]  = *(const uint4*)(arow + kc0 + kk + hi * 16);
            a[s].u[1]  = *(const uint4*)(arow + kc0 + kk + hi * 16 + 8);
            // B frags: column = lane&15 (+16 for subtile 1), 16 contiguous K
            b0[s].u[0] = *(const uint4*)(&Bs[col][kk + hi * 16]);
            b0[s].u[1] = *(const uint4*)(&Bs[col][kk + hi * 16 + 8]);
            b1[s].u[0] = *(const uint4*)(&Bs[16 + col][kk + hi * 16]);
            b1[s].u[1] = *(const uint4*)(&Bs[16 + col][kk + hi * 16 + 8]);
        }
#pragma unroll
        for (int s = 0; s < 3; ++s) {
            acc0 = __builtin_amdgcn_wmma_f32_16x16x32_bf16(
                       false, a[s].v, false, b0[s].v, (short)0, acc0, false, false);
            acc1 = __builtin_amdgcn_wmma_f32_16x16x32_bf16(
                       false, a[s].v, false, b1[s].v, (short)0, acc1, false, false);
        }
    }

    const float bias0 = conv_b[n0 + col];
    const float bias1 = conv_b[n0 + 16 + col];
#pragma unroll
    for (int r = 0; r < 8; ++r) {
        int gr = wave * 16 + hi * 8 + r;         // C layout: VGPR r -> row r + 8*hi
        float v0 = acc0[r] + bias0;
        float v1 = acc1[r] + bias1;
        c[(size_t)gr * Ntot + n0 + col]      = v0 > 0.f ? v0 : 0.f;
        c[(size_t)gr * Ntot + n0 + 16 + col] = v1 > 0.f ? v1 : 0.f;
    }
}

// ---------------- 3) res1[b,h,t] = sum_k c[b,h,k,k+t] ---------------------
__global__ void res1_kernel(const float* __restrict__ c, float* __restrict__ res1) {
    int i = blockIdx.x * blockDim.x + threadIdx.x;
    if (i >= Bx * HIDC * Lt) return;
    int b = i / (HIDC * Lt);
    int r = i - b * (HIDC * Lt);
    int h = r / Lt;
    int t = r - h * Lt;
    const float* base = c + (size_t)b * Ntot + (size_t)h * (CK * Pn);
    float s = 0.f;
#pragma unroll
    for (int k = 0; k < CK; ++k) s += base[k * Pn + k + t];
    res1[i] = s;
}

// ---------------- 4) gi1[t,b,g]: input-gate precompute for GRU-1 ----------
__global__ void gi1_kernel(const float* __restrict__ res1,
                           const float* __restrict__ wih,
                           const float* __restrict__ bih,
                           float* __restrict__ gi1) {
    int i = blockIdx.x * blockDim.x + threadIdx.x;
    if (i >= Lt * Bx * 3 * HIDR) return;
    int t = i / (Bx * 3 * HIDR);
    int r = i - t * (Bx * 3 * HIDR);
    int b = r / (3 * HIDR);
    int g = r - b * (3 * HIDR);
    const float* rb = res1 + (size_t)b * (HIDC * Lt) + t;   // stride Lt over ch
    const float* w  = wih + g * HIDC;
    float s = bih[g];
#pragma unroll 5
    for (int ch = 0; ch < HIDC; ++ch) s += rb[ch * Lt] * w[ch];
    gi1[i] = s;
}

// ---------------- 5) gis[pt,bb,g]: input-gate precompute for skip GRU -----
__global__ void gis_kernel(const float* __restrict__ res1,
                           const float* __restrict__ wih,
                           const float* __restrict__ bih,
                           float* __restrict__ gis) {
    int i = blockIdx.x * blockDim.x + threadIdx.x;
    if (i >= PT * (Bx * SKIP) * 3 * HIDS) return;
    int pt = i / ((Bx * SKIP) * 3 * HIDS);
    int r  = i - pt * ((Bx * SKIP) * 3 * HIDS);
    int bb = r / (3 * HIDS);
    int g  = r - bb * (3 * HIDS);
    int b  = bb / SKIP;
    int sk = bb - b * SKIP;
    int t  = (Lt - PT * SKIP) + pt * SKIP + sk;             // 19 + pt*24 + sk
    const float* rb = res1 + (size_t)b * (HIDC * Lt) + t;
    const float* w  = wih + g * HIDC;
    float s = bih[g];
#pragma unroll 5
    for (int ch = 0; ch < HIDC; ++ch) s += rb[ch * Lt] * w[ch];
    gis[i] = s;
}

__device__ __forceinline__ float sigf(float v) { return 1.f / (1.f + __expf(-v)); }

// ---------------- 6) GRU-1 sequential loop (batch split over 4 WGs) -------
#define G1B 32   // batches per workgroup
__global__ void __launch_bounds__(256)
gru1_kernel(const float* __restrict__ gi1,
            const float* __restrict__ whh_g,
            const float* __restrict__ bhh_g,
            float* __restrict__ hout) {
    __shared__ float hbuf[2][G1B][HIDR];
    __shared__ float whh[3 * HIDR][HIDR];
    __shared__ float bhh[3 * HIDR];
    const int bstart = blockIdx.x * G1B;

    for (int i = threadIdx.x; i < 3 * HIDR * HIDR; i += 256)
        whh[i / HIDR][i % HIDR] = whh_g[i];
    for (int i = threadIdx.x; i < 3 * HIDR; i += 256) bhh[i] = bhh_g[i];
    for (int i = threadIdx.x; i < G1B * HIDR; i += 256)
        hbuf[0][i / HIDR][i % HIDR] = 0.f;
    __syncthreads();

    for (int t = 0; t < Lt; ++t) {
        int p = t & 1;
        for (int i = threadIdx.x; i < G1B * HIDR; i += 256) {
            int bl = i / HIDR;
            int j  = i - bl * HIDR;
            const float* h = hbuf[p][bl];
            float gr = bhh[j], gz = bhh[HIDR + j], gn = bhh[2 * HIDR + j];
            const float* wr = whh[j];
            const float* wz = whh[HIDR + j];
            const float* wn = whh[2 * HIDR + j];
#pragma unroll 5
            for (int kk = 0; kk < HIDR; ++kk) {
                float hv = h[kk];
                gr += hv * wr[kk]; gz += hv * wz[kk]; gn += hv * wn[kk];
            }
            const float* gi = gi1 + ((size_t)t * Bx + bstart + bl) * (3 * HIDR);
            float rg = sigf(gi[j] + gr);
            float zg = sigf(gi[HIDR + j] + gz);
            float ng = tanhf(gi[2 * HIDR + j] + rg * gn);
            hbuf[1 - p][bl][j] = (1.f - zg) * ng + zg * h[j];
        }
        __syncthreads();
    }
    for (int i = threadIdx.x; i < G1B * HIDR; i += 256)
        hout[(size_t)(bstart + i / HIDR) * HIDR + (i % HIDR)] = hbuf[Lt & 1][i / HIDR][i % HIDR];
}

// ---------------- 7) skip GRU sequential loop (3072 batches / 12 WGs) -----
#define GSB 256  // batches per workgroup
__global__ void __launch_bounds__(256)
grus_kernel(const float* __restrict__ gis,
            const float* __restrict__ whh_g,
            const float* __restrict__ bhh_g,
            float* __restrict__ hout) {
    __shared__ float hbuf[2][GSB][HIDS];
    __shared__ float whh[3 * HIDS][HIDS];
    __shared__ float bhh[3 * HIDS];
    const int bstart = blockIdx.x * GSB;

    for (int i = threadIdx.x; i < 3 * HIDS * HIDS; i += 256)
        whh[i / HIDS][i % HIDS] = whh_g[i];
    for (int i = threadIdx.x; i < 3 * HIDS; i += 256) bhh[i] = bhh_g[i];
    for (int i = threadIdx.x; i < GSB * HIDS; i += 256)
        hbuf[0][i / HIDS][i % HIDS] = 0.f;
    __syncthreads();

    for (int t = 0; t < PT; ++t) {
        int p = t & 1;
        for (int i = threadIdx.x; i < GSB * HIDS; i += 256) {
            int bl = i / HIDS;
            int j  = i - bl * HIDS;
            const float* h = hbuf[p][bl];
            float gr = bhh[j], gz = bhh[HIDS + j], gn = bhh[2 * HIDS + j];
#pragma unroll
            for (int kk = 0; kk < HIDS; ++kk) {
                float hv = h[kk];
                gr += hv * whh[j][kk];
                gz += hv * whh[HIDS + j][kk];
                gn += hv * whh[2 * HIDS + j][kk];
            }
            const float* gi = gis + ((size_t)t * (Bx * SKIP) + bstart + bl) * (3 * HIDS);
            float rg = sigf(gi[j] + gr);
            float zg = sigf(gi[HIDS + j] + gz);
            float ng = tanhf(gi[2 * HIDS + j] + rg * gn);
            hbuf[1 - p][bl][j] = (1.f - zg) * ng + zg * h[j];
        }
        __syncthreads();
    }
    for (int i = threadIdx.x; i < GSB * HIDS; i += 256)
        hout[(size_t)(bstart + i / HIDS) * HIDS + (i % HIDS)] = hbuf[PT & 1][i / HIDS][i % HIDS];
}

// ---------------- 8) final linear + highway + sigmoid ---------------------
__global__ void final_kernel(const float* __restrict__ h1,
                             const float* __restrict__ hs,
                             const float* __restrict__ x,
                             const float* __restrict__ l1_w,
                             const float* __restrict__ l1_b,
                             const float* __restrict__ hw_w,
                             const float* __restrict__ hw_b,
                             float* __restrict__ out) {
    int i = blockIdx.x * blockDim.x + threadIdx.x;
    if (i >= Bx * Mout) return;
    int b = i / Mout, m = i - b * Mout;
    const float* w = l1_w + m * (HIDR + SKIP * HIDS);
    float s = l1_b[m];
#pragma unroll 5
    for (int k = 0; k < HIDR; ++k) s += h1[b * HIDR + k] * w[k];
    for (int sk = 0; sk < SKIP; ++sk)
#pragma unroll
        for (int j = 0; j < HIDS; ++j)
            s += hs[(b * SKIP + sk) * HIDS + j] * w[HIDR + sk * HIDS + j];
    float z = hw_b[0];
#pragma unroll
    for (int wv = 0; wv < HW; ++wv)
        z += x[(size_t)b * Kdim + (Pn - HW + wv) * Mout + m] * hw_w[wv];
    out[i] = sigf(s + z);
}

// ---------------- host launch ---------------------------------------------
extern "C" void kernel_launch(void* const* d_in, const int* in_sizes, int n_in,
                              void* d_out, int out_size, void* d_ws, size_t ws_size,
                              hipStream_t stream) {
    const float* x      = (const float*)d_in[0];
    const float* conv_w = (const float*)d_in[1];
    const float* conv_b = (const float*)d_in[2];
    const float* g1_wih = (const float*)d_in[3];
    const float* g1_whh = (const float*)d_in[4];
    const float* g1_bih = (const float*)d_in[5];
    const float* g1_bhh = (const float*)d_in[6];
    const float* gs_wih = (const float*)d_in[7];
    const float* gs_whh = (const float*)d_in[8];
    const float* gs_bih = (const float*)d_in[9];
    const float* gs_bhh = (const float*)d_in[10];
    const float* l1_w   = (const float*)d_in[11];
    const float* l1_b   = (const float*)d_in[12];
    const float* hw_w   = (const float*)d_in[13];
    const float* hw_b   = (const float*)d_in[14];
    float* out = (float*)d_out;

    char* ws = (char*)d_ws;
    size_t off = 0;
    auto alloc = [&](size_t bytes) -> void* {
        void* p = (void*)(ws + off);
        off += (bytes + 255) & ~(size_t)255;
        return p;
    };
    __bf16* xbf  = (__bf16*)alloc((size_t)Bx * Kdim * sizeof(__bf16));
    float*  c    = (float*) alloc((size_t)Bx * Ntot * sizeof(float));
    float*  res1 = (float*) alloc((size_t)Bx * HIDC * Lt * sizeof(float));
    float*  gi1  = (float*) alloc((size_t)Lt * Bx * 3 * HIDR * sizeof(float));
    float*  gis  = (float*) alloc((size_t)PT * Bx * SKIP * 3 * HIDS * sizeof(float));
    float*  h1   = (float*) alloc((size_t)Bx * HIDR * sizeof(float));
    float*  hs   = (float*) alloc((size_t)Bx * SKIP * HIDS * sizeof(float));

    cvt_x_kernel<<<(Bx * Kdim + 255) / 256, 256, 0, stream>>>(x, xbf);
    conv_gemm_kernel<<<Ntot / NT, 256, 0, stream>>>(xbf, conv_w, conv_b, c);
    res1_kernel<<<(Bx * HIDC * Lt + 255) / 256, 256, 0, stream>>>(c, res1);
    gi1_kernel<<<((size_t)Lt * Bx * 3 * HIDR + 255) / 256, 256, 0, stream>>>(res1, g1_wih, g1_bih, gi1);
    gis_kernel<<<((size_t)PT * Bx * SKIP * 3 * HIDS + 255) / 256, 256, 0, stream>>>(res1, gs_wih, gs_bih, gis);
    gru1_kernel<<<Bx / G1B, 256, 0, stream>>>(gi1, g1_whh, g1_bhh, h1);
    grus_kernel<<<(Bx * SKIP) / GSB, 256, 0, stream>>>(gis, gs_whh, gs_bhh, hs);
    final_kernel<<<(Bx * Mout + 255) / 256, 256, 0, stream>>>(h1, hs, x, l1_w, l1_b, hw_w, hw_b, out);
}
